// Head_1005022347878
// MI455X (gfx1250) — compile-verified, matrix operand order checked by reading, not run on hardware
//
#include <hip/hip_runtime.h>
#include <hip/hip_bf16.h>

typedef __attribute__((ext_vector_type(16))) _Float16 v16h;
typedef __attribute__((ext_vector_type(4)))  _Float16 v4h;
typedef __attribute__((ext_vector_type(8)))  float    v8f;
typedef __attribute__((ext_vector_type(4)))  unsigned u32x4;
typedef __attribute__((ext_vector_type(8)))  int      i32x8;
typedef __attribute__((ext_vector_type(4)))  int      i32x4;

constexpr int Bn = 8;      // batch
constexpr int Tn = 2048;   // sequence length
constexpr int Dn = 1024;   // embedding dim
constexpr int Hn = 64;     // head size

#ifndef __has_builtin
#define __has_builtin(x) 0
#endif
#if __has_builtin(__builtin_amdgcn_tensor_load_to_lds) && \
    __has_builtin(__builtin_amdgcn_s_wait_tensorcnt)
#define USE_TDM 1
#else
#define USE_TDM 0
#endif

// ---- WMMA 16x16x32 f16 fragment index helpers (CDNA5 ISA 7.12.2) ----------
// A (16x32, MxK): lane holds row M = lane%16; half h -> K:
//   lanes 0-15 : h<8 -> K=h   ; h>=8 -> K=h+8
//   lanes 16-31: h<8 -> K=h+8 ; h>=8 -> K=h+16
// B (32x16, KxN): symmetric, lane holds col N = lane%16, same K map.
// C/D (16x16 f32): lane holds col N = lane%16; vgpr r -> row M = r + 8*(lane/16).
__device__ __forceinline__ int kmap_h(int lane, int h) {
  int k = (h < 8) ? h : h + 8;
  return k + ((lane & 16) ? 8 : 0);
}
__device__ __forceinline__ int kmap_p(int lane, int p) {  // halves 2p,2p+1
  int k = (p < 4) ? 2 * p : 2 * p + 8;
  return k + ((lane & 16) ? 8 : 0);
}

// Load an A/B fragment from a row-major f16 matrix (rows = M/N, cols = K).
__device__ __forceinline__ v16h load_frag_f16(const _Float16* base, int row0,
                                              int ld, int k0, int lane) {
  union { v16h v; unsigned u[8]; } t;
  const _Float16* p = base + (size_t)(row0 + (lane & 15)) * ld + k0;
#pragma unroll
  for (int i = 0; i < 8; ++i)
    t.u[i] = *(const unsigned*)(p + kmap_p(lane, i));
  return t.v;
}

// Build an A fragment from a row-major 16x32 fp32 tile (e.g. TDM-staged LDS).
__device__ __forceinline__ v16h frag_from_f32(const float* base, int lane) {
  v16h a;
#pragma unroll
  for (int p = 0; p < 8; ++p) {
    const float2 f =
        *(const float2*)(base + (size_t)(lane & 15) * 32 + kmap_p(lane, p));
    a[2 * p]     = (_Float16)f.x;
    a[2 * p + 1] = (_Float16)f.y;
  }
  return a;
}

__device__ __forceinline__ v8f wmma_f16(v16h a, v16h b, v8f c) {
  return __builtin_amdgcn_wmma_f32_16x16x32_f16(false, a, false, b, (short)0, c,
                                                false, false);
}

// ---------------------------------------------------------------------------
// TDM: async-DMA a 2-D fp32 tile (tile_w x tile_h, row stride in elements)
// from global memory into LDS.  D# packing per CDNA5 ISA 8.3/8.4:
//  G0: count=1 | lds_addr | global_addr[56:0] | type=2
//  G1: data_size=4B | tensor_dim0=D | tensor_dim1=T | tile dims | dim0 stride
//  G2/G3: zero (<=2D tensor).  Completion tracked by TENSORcnt.
// ---------------------------------------------------------------------------
__device__ __forceinline__ void tdm_load_tile_f32(unsigned lds_off,
                                                  const float* gsrc,
                                                  unsigned tile_w,
                                                  unsigned tile_h,
                                                  unsigned row_stride) {
#if USE_TDM
  const unsigned long long ga = (unsigned long long)(uintptr_t)gsrc;
  u32x4 g0;
  g0.x = 1u;                                                 // count=1
  g0.y = lds_off;                                            // lds_addr (bytes)
  g0.z = (unsigned)ga;                                       // addr[31:0]
  g0.w = ((unsigned)(ga >> 32) & 0x01ffffffu) | (2u << 30);  // addr[56:32]|type
  i32x8 g1;
  g1.s0 = (int)(2u << 16);                            // data_size = 4B
  g1.s1 = (int)((unsigned)Dn << 16);                  // tensor_dim0 lo16
  g1.s2 = (int)(((unsigned)Dn >> 16) | ((unsigned)Tn << 16));  // d0 hi | d1 lo
  g1.s3 = (int)(((unsigned)Tn >> 16) | (tile_w << 16));        // d1 hi | tile0
  g1.s4 = (int)tile_h;                                // tile_dim1 | tile_dim2=0
  g1.s5 = (int)row_stride;                            // dim0_stride[31:0]
  g1.s6 = 0;                                          // stride hi | d1_stride lo
  g1.s7 = 0;
  const i32x4 gz4 = (i32x4)0;
#if __clang_major__ >= 23
  const i32x8 gz8 = (i32x8)0;
  __builtin_amdgcn_tensor_load_to_lds(g0, g1, gz4, gz4, gz8, 0);
#else
  __builtin_amdgcn_tensor_load_to_lds(g0, g1, gz4, gz4, 0);
#endif
#else
  (void)lds_off; (void)gsrc; (void)tile_w; (void)tile_h; (void)row_stride;
#endif
}

// ---------------------------------------------------------------------------
// Kernel 0: pack W{q,k,v} (fp32 [D][64]) into fragment-native f16 layout:
// one aligned 32B load per lane per fragment in the hot loop, L2-resident.
// wp index: (((w*(D/32) + kb)*4 + nt)*32 + lane)*16 + h
// ---------------------------------------------------------------------------
__global__ void __launch_bounds__(32)
pack_w_kernel(const float* __restrict__ Wq, const float* __restrict__ Wk,
              const float* __restrict__ Wv, _Float16* __restrict__ wp) {
  const int lane = threadIdx.x;
  const int kb = blockIdx.x;  // k0 = kb*32
  const int w  = blockIdx.y;
  const float* W = (w == 0) ? Wq : (w == 1) ? Wk : Wv;
  const int col = lane & 15;
#pragma unroll
  for (int nt = 0; nt < 4; ++nt) {
    _Float16* dst = wp + (((size_t)w * (Dn / 32) + kb) * 4 + nt) * 512 + lane * 16;
#pragma unroll
    for (int h = 0; h < 16; ++h)
      dst[h] = (_Float16)W[(size_t)(kb * 32 + kmap_h(lane, h)) * Hn + nt * 16 + col];
  }
}

// ---------------------------------------------------------------------------
// Kernel 1: fused Q/K/V projection. Context read ONCE; the 16x32 fp32 slab is
// DMA'd into double-buffered LDS by the Tensor Data Mover (pipelined one step
// ahead, s_wait_tensorcnt), then one A fragment feeds 12 WMMAs per k-step.
// Q,K stored row-major f16 [B*T][64]; V stored transposed f16 [B][64][T].
// ---------------------------------------------------------------------------
__global__ void __launch_bounds__(32)
qkv_proj_kernel(const float* __restrict__ ctx, const _Float16* __restrict__ wp,
                _Float16* __restrict__ qf, _Float16* __restrict__ kf,
                _Float16* __restrict__ vt) {
  const int lane = threadIdx.x;
  const int qt = blockIdx.x, b = blockIdx.y;
  const int q0 = qt * 16;
  const int col = lane & 15;
  const int hi  = (lane >> 4) & 1;
  const float* cbase = ctx + ((size_t)b * Tn + q0) * Dn;

  v8f acc[3][4] = {};

#if USE_TDM
  __shared__ float As[2][16 * 32];  // double-buffered fp32 slabs (2x2KB)
  const unsigned lds_off[2] = {(unsigned)(uintptr_t)&As[0][0],
                               (unsigned)(uintptr_t)&As[1][0]};
  tdm_load_tile_f32(lds_off[0], cbase, 32, 16, Dn);  // prime the pipeline
#else
  __shared__ _Float16 At[16 * 32];
#endif

  for (int k0 = 0; k0 < Dn; k0 += 32) {
#if USE_TDM
    const int pb = (k0 >> 5) & 1;
    if (k0 + 32 < Dn) {
      tdm_load_tile_f32(lds_off[pb ^ 1], cbase + k0 + 32, 32, 16, Dn);
      __builtin_amdgcn_s_wait_tensorcnt(1);  // current slab landed
    } else {
      __builtin_amdgcn_s_wait_tensorcnt(0);
    }
    __syncthreads();
    const v16h a = frag_from_f32(&As[pb][0], lane);
#else
    // ---- manual staging: 16x32 fp32 slab -> f16 LDS, coalesced b128 ----
#pragma unroll
    for (int i = 0; i < 4; ++i) {
      const int f = i * 32 + lane;      // flat float4 index in tile
      const int r = f >> 3, c = f & 7;  // row, float4-column
      const float4 v = *(const float4*)(cbase + (size_t)r * Dn + k0 + c * 4);
      v4h h4; h4.x = (_Float16)v.x; h4.y = (_Float16)v.y;
      h4.z = (_Float16)v.z; h4.w = (_Float16)v.w;
      *(v4h*)(At + r * 32 + c * 4) = h4;
    }
    if (k0 + 32 < Dn)
      __builtin_prefetch(cbase + (size_t)col * Dn + k0 + 32 + hi * 16, 0, 3);
    __syncthreads();
    const v16h a = load_frag_f16(At, 0, 32, 0, lane);
#endif

    const size_t kb = (size_t)(k0 >> 5);
#pragma unroll
    for (int w = 0; w < 3; ++w) {
#pragma unroll
      for (int nt = 0; nt < 4; ++nt) {
        const v16h bw = *(const v16h*)(
            wp + (((size_t)w * (Dn / 32) + kb) * 4 + nt) * 512 + lane * 16);
        acc[w][nt] = wmma_f16(a, bw, acc[w][nt]);
      }
    }
    __syncthreads();
  }

  // ---- epilogue: Q,K row-major; V transposed [b][h][t] ----
#pragma unroll
  for (int nt = 0; nt < 4; ++nt) {
#pragma unroll
    for (int r = 0; r < 8; ++r) {
      const size_t rq = ((size_t)b * Tn + q0 + r + 8 * hi) * Hn + nt * 16 + col;
      qf[rq] = (_Float16)acc[0][nt][r];
      kf[rq] = (_Float16)acc[1][nt][r];
      vt[((size_t)b * Hn + nt * 16 + col) * Tn + q0 + r + 8 * hi] =
          (_Float16)acc[2][nt][r];
    }
  }
}

// ---------------------------------------------------------------------------
// Kernel 2: causal flash attention. One wave owns a 16-row query tile;
// iterates 32-key tiles up to the diagonal with online softmax.
// ---------------------------------------------------------------------------
__global__ void __launch_bounds__(32)
flash_attn_kernel(const _Float16* __restrict__ qf,
                  const _Float16* __restrict__ kf,
                  const _Float16* __restrict__ vt, float* __restrict__ out) {
  __shared__ _Float16 Pb[16 * 32];  // P tile staging: D-layout -> A-layout

  const int lane = threadIdx.x;
  const int qt = blockIdx.x, b = blockIdx.y;
  const int q0 = qt * 16;
  const int col = lane & 15;
  const int hi  = (lane >> 4) & 1;

  const _Float16* Qb = qf + (size_t)b * Tn * Hn;
  const _Float16* Kb = kf + (size_t)b * Tn * Hn;
  const _Float16* Vb = vt + (size_t)b * Hn * Tn;

  const v16h qa0 = load_frag_f16(Qb, q0, Hn, 0, lane);   // hs 0..31
  const v16h qa1 = load_frag_f16(Qb, q0, Hn, 32, lane);  // hs 32..63

  v8f o[4] = {};
  float mi[8], li[8];
#pragma unroll
  for (int r = 0; r < 8; ++r) { mi[r] = -3.0e38f; li[r] = 0.f; }

  const float scale = 0.125f;  // 64^-0.5
  const int ktmax = (q0 + 15) >> 5;
  for (int kt = 0; kt <= ktmax; ++kt) {
    const int kb = kt * 32;
    if (kt < ktmax)  // prefetch next K tile rows (128B/row covers hs 0..63)
      __builtin_prefetch(Kb + (size_t)(kb + 32 + lane) * Hn, 0, 3);

    // S = Q * K^T : two 16x16 column blocks, K-dim = 64 (2 WMMAs each)
    v8f s0 = {}, s1 = {};
    s0 = wmma_f16(qa0, load_frag_f16(Kb, kb, Hn, 0, lane), s0);
    s0 = wmma_f16(qa1, load_frag_f16(Kb, kb, Hn, 32, lane), s0);
    s1 = wmma_f16(qa0, load_frag_f16(Kb, kb + 16, Hn, 0, lane), s1);
    s1 = wmma_f16(qa1, load_frag_f16(Kb, kb + 16, Hn, 32, lane), s1);

    const bool need_mask = (kb + 31 > q0);
#pragma unroll
    for (int r = 0; r < 8; ++r) {
      const int mq = q0 + r + 8 * hi;  // absolute query index of this row
      float a0 = s0[r] * scale;
      float a1 = s1[r] * scale;
      if (need_mask) {
        if (kb + col > mq)      a0 = -3.0e38f;
        if (kb + 16 + col > mq) a1 = -3.0e38f;
      }
      s0[r] = a0; s1[r] = a1;
    }

    // Online softmax: each row lives in one VGPR across a 16-lane half-wave.
#pragma unroll
    for (int r = 0; r < 8; ++r) {
      float rm = fmaxf(s0[r], s1[r]);
#pragma unroll
      for (int m = 8; m >= 1; m >>= 1) rm = fmaxf(rm, __shfl_xor(rm, m, 32));
      const float mn = fmaxf(mi[r], rm);
      const float p0 = __expf(s0[r] - mn);
      const float p1 = __expf(s1[r] - mn);
      float rs = p0 + p1;
#pragma unroll
      for (int m = 8; m >= 1; m >>= 1) rs += __shfl_xor(rs, m, 32);
      const float alpha = __expf(mi[r] - mn);
      li[r] = li[r] * alpha + rs;
      mi[r] = mn;
#pragma unroll
      for (int nt = 0; nt < 4; ++nt) o[nt][r] *= alpha;
      Pb[(r + 8 * hi) * 32 + col]      = (_Float16)p0;
      Pb[(r + 8 * hi) * 32 + 16 + col] = (_Float16)p1;
    }
    __syncthreads();
    const v16h pa = load_frag_f16(Pb, 0, 32, 0, lane);  // P as A fragment
    __syncthreads();

    // O += P * V  (V^T stored [h][t]: B fragment rows = h, K = key index)
#pragma unroll
    for (int nt = 0; nt < 4; ++nt) {
      const v16h vb = load_frag_f16(Vb, nt * 16, Tn, kb, lane);
      o[nt] = wmma_f16(pa, vb, o[nt]);
    }
  }

  // epilogue: normalize and store fp32
#pragma unroll
  for (int nt = 0; nt < 4; ++nt)
#pragma unroll
    for (int r = 0; r < 8; ++r)
      out[((size_t)b * Tn + q0 + r + 8 * hi) * Hn + nt * 16 + col] =
          o[nt][r] / li[r];
}

extern "C" void kernel_launch(void* const* d_in, const int* in_sizes, int n_in,
                              void* d_out, int out_size, void* d_ws,
                              size_t ws_size, hipStream_t stream) {
  (void)in_sizes; (void)n_in; (void)out_size; (void)ws_size;
  const float* ctx = (const float*)d_in[0];
  const float* Wq  = (const float*)d_in[1];
  const float* Wk  = (const float*)d_in[2];
  const float* Wv  = (const float*)d_in[3];
  float* out = (float*)d_out;

  // workspace: Qf16 | Kf16 | Vt_f16 | packed weights  = 6 MB + 384 KB
  const size_t bth = (size_t)Bn * Tn * Hn;
  _Float16* qf  = (_Float16*)d_ws;
  _Float16* kf  = qf + bth;
  _Float16* vt  = kf + bth;
  _Float16* wpk = vt + bth;

  pack_w_kernel<<<dim3(Dn / 32, 3), 32, 0, stream>>>(Wq, Wk, Wv, wpk);
  qkv_proj_kernel<<<dim3(Tn / 16, Bn), 32, 0, stream>>>(ctx, wpk, qf, kf, vt);
  flash_attn_kernel<<<dim3(Tn / 16, Bn), 32, 0, stream>>>(qf, kf, vt, out);
}
